// VQVAE_60507499266848
// MI455X (gfx1250) — matrix operand order repaired and checked by reading, not compiled, not run on hardware
//
#include <hip/hip_runtime.h>
#include <math.h>

// ---------------- CDNA5 WMMA types ----------------
typedef __attribute__((ext_vector_type(16))) _Float16 v16h;
typedef __attribute__((ext_vector_type(8)))  float    v8f;
typedef __attribute__((ext_vector_type(4)))  unsigned int u32x4;

// Packed B layout (shared by both GEMMs): for ntile nt (16 cols), ktile kt (32 rows):
//   Bp[((nt*(K/32)+kt)*32 + lane)*16 + j]
//   lane<16 : col = nt*16+lane,    kin = j       (K rows 0..15 of the tile)
//   lane>=16: col = nt*16+lane-16, kin = 16+j    (K rows 16..31)

// =====================================================================
// Wide WMMA GEMM for N % 64 == 0:
//   block tile 128x64, 8 waves, wave tile 32x32 (4 accumulators).
//   A rows are streamed once per 64 output columns (A-traffic halved vs
//   the 32-col kernel); B read directly from packed global (L2 resident).
// grid = (M/128, N/64).  K%32==0, M%128==0.
// =====================================================================
__global__ __launch_bounds__(256)
void k_gemm_wmma_n64(const _Float16* __restrict__ A, const _Float16* __restrict__ Bp,
                     float* __restrict__ C, int M, int N, int K) {
  const int tid  = threadIdx.x;
  const int lane = tid & 31;
  const int wave = tid >> 5;
  const int ktiles = K >> 5;
  const size_t halvesPerNt = (size_t)K * 16;

  const int wm = wave & 3;                     // 4 M-subtiles of 32 rows
  const int wn = wave >> 2;                    // 2 N-subtiles of 32 cols
  const size_t m0 = (size_t)blockIdx.x * 128 + (size_t)wm * 32;
  const int r    = lane & 15;
  const int koff = (lane >> 4) << 4;           // 0 or 16 (contiguous k, matches packed B)

  v8f acc00 = {}, acc01 = {}, acc10 = {}, acc11 = {};
  const _Float16* arow0 = A + (m0 + r) * (size_t)K;
  const _Float16* arow1 = A + (m0 + 16 + r) * (size_t)K;
  const int nt0 = blockIdx.y * 4 + wn * 2;
  const _Float16* bb0 = Bp + (size_t)nt0 * halvesPerNt + lane * 16;
  const _Float16* bb1 = bb0 + halvesPerNt;

  for (int kt = 0; kt < ktiles; ++kt) {
    const int k0 = kt << 5;
    union { v16h v; u32x4 q[2]; } a0, a1, b0, b1;
    a0.q[0] = *(const u32x4*)(arow0 + k0 + koff);
    a0.q[1] = *(const u32x4*)(arow0 + k0 + koff + 8);
    a1.q[0] = *(const u32x4*)(arow1 + k0 + koff);
    a1.q[1] = *(const u32x4*)(arow1 + k0 + koff + 8);
    b0.q[0] = *(const u32x4*)(bb0 + (size_t)kt * 512);
    b0.q[1] = *(const u32x4*)(bb0 + (size_t)kt * 512 + 8);
    b1.q[0] = *(const u32x4*)(bb1 + (size_t)kt * 512);
    b1.q[1] = *(const u32x4*)(bb1 + (size_t)kt * 512 + 8);
    __builtin_prefetch(arow0 + k0 + 128, 0, 3);   // global_prefetch_b8, near cache
    acc00 = __builtin_amdgcn_wmma_f32_16x16x32_f16(false, a0.v, false, b0.v,
                                                   (short)0, acc00, false, false);
    acc01 = __builtin_amdgcn_wmma_f32_16x16x32_f16(false, a0.v, false, b1.v,
                                                   (short)0, acc01, false, false);
    acc10 = __builtin_amdgcn_wmma_f32_16x16x32_f16(false, a1.v, false, b0.v,
                                                   (short)0, acc10, false, false);
    acc11 = __builtin_amdgcn_wmma_f32_16x16x32_f16(false, a1.v, false, b1.v,
                                                   (short)0, acc11, false, false);
  }

  // C/D layout: VGPR i -> row = i + 8*(lane>=16), col = lane&15
  const int colb  = blockIdx.y * 64 + wn * 32 + (lane & 15);
  const int rbase = (lane >> 4) << 3;
  for (int i = 0; i < 8; ++i) {
    C[(m0 + rbase + i) * (size_t)N + colb]           = acc00[i];
    C[(m0 + rbase + i) * (size_t)N + colb + 16]      = acc01[i];
    C[(m0 + 16 + rbase + i) * (size_t)N + colb]      = acc10[i];
    C[(m0 + 16 + rbase + i) * (size_t)N + colb + 16] = acc11[i];
  }
}

// =====================================================================
// Narrow WMMA GEMM (N % 32 == 0), LDS-staged B: block tile 128x32,
// 8 waves, wave tile 32x16.  Used for the N=32 resblock-2 GEMMs.
// grid = (M/128, N/32).
// =====================================================================
__global__ __launch_bounds__(256)
void k_gemm_wmma(const _Float16* __restrict__ A, const _Float16* __restrict__ Bp,
                 float* __restrict__ C, int M, int N, int K) {
  __shared__ _Float16 sB[576 * 32];            // up to K=576, 32 cols = 36 KB
  const int tid  = threadIdx.x;
  const int lane = tid & 31;
  const int wave = tid >> 5;
  const int ktiles = K >> 5;
  const int halvesPerNt = K * 16;
  const int nt0 = blockIdx.y * 2;

  const int totalVec = (halvesPerNt * 2) >> 3;
  for (int i = tid; i < totalVec; i += 256) {
    int h   = i << 3;
    int lnt = h / halvesPerNt;
    int off = h - lnt * halvesPerNt;
    *(u32x4*)(sB + lnt * halvesPerNt + off) =
        *(const u32x4*)(Bp + (size_t)(nt0 + lnt) * halvesPerNt + off);
  }
  __syncthreads();

  const int wm  = wave & 3;
  const int lnt = wave >> 2;
  const size_t m0 = (size_t)blockIdx.x * 128 + (size_t)wm * 32;
  const int r    = lane & 15;
  const int koff = (lane >> 4) << 4;

  v8f acc0 = {}; v8f acc1 = {};
  const _Float16* arow0 = A + (m0 + r) * (size_t)K;
  const _Float16* arow1 = A + (m0 + 16 + r) * (size_t)K;
  const _Float16* bbase = sB + lnt * halvesPerNt + lane * 16;

  for (int kt = 0; kt < ktiles; ++kt) {
    const int k0 = kt << 5;
    union { v16h v; u32x4 q[2]; } a0, a1, b;
    a0.q[0] = *(const u32x4*)(arow0 + k0 + koff);
    a0.q[1] = *(const u32x4*)(arow0 + k0 + koff + 8);
    a1.q[0] = *(const u32x4*)(arow1 + k0 + koff);
    a1.q[1] = *(const u32x4*)(arow1 + k0 + koff + 8);
    const _Float16* bp = bbase + kt * 512;
    b.q[0] = *(const u32x4*)(bp);
    b.q[1] = *(const u32x4*)(bp + 8);
    __builtin_prefetch(arow0 + k0 + 128, 0, 3);
    acc0 = __builtin_amdgcn_wmma_f32_16x16x32_f16(false, a0.v, false, b.v,
                                                  (short)0, acc0, false, false);
    acc1 = __builtin_amdgcn_wmma_f32_16x16x32_f16(false, a1.v, false, b.v,
                                                  (short)0, acc1, false, false);
  }

  const int col   = blockIdx.y * 32 + lnt * 16 + (lane & 15);
  const int rbase = (lane >> 4) << 3;
  for (int i = 0; i < 8; ++i) {
    C[(m0 + rbase + i) * (size_t)N + col]      = acc0[i];
    C[(m0 + 16 + rbase + i) * (size_t)N + col] = acc1[i];
  }
}

// ---------------- im2col: NHWC f32 -> A f16 [M, K=kh*kw*C] ----------------
__global__ void k_im2col(const float* __restrict__ x, _Float16* __restrict__ A,
                         int Bn, int IH, int IW, int C, int kh, int kw,
                         int stride, int pad, int OH, int OW) {
  const int taps = kh * kw;
  const size_t total = (size_t)Bn * OH * OW * taps;
  size_t t = (size_t)blockIdx.x * 256 + threadIdx.x;
  if (t >= total) return;
  int tap = (int)(t % taps);
  size_t m = t / taps;
  int ox = (int)(m % OW);
  size_t tmp = m / OW;
  int oy = (int)(tmp % OH);
  int b  = (int)(tmp / OH);
  int ky = tap / kw, kx = tap % kw;
  int iy = oy * stride - pad + ky;
  int ix = ox * stride - pad + kx;
  const int K = taps * C;
  _Float16* dst = A + m * (size_t)K + (size_t)tap * C;
  if (iy < 0 || iy >= IH || ix < 0 || ix >= IW) {
    for (int c = 0; c < C; ++c) dst[c] = (_Float16)0.0f;
  } else {
    const float* src = x + (((size_t)b * IH + iy) * IW + ix) * C;
    for (int c = 0; c < C; ++c) dst[c] = (_Float16)src[c];
  }
}

// ------------- weight pack into WMMA-friendly B layout -------------
// mode 0: conv OIHW  w[Co][Ci][kh][kw],  B[k][co], k=(ky*kw+kx)*Ci+ci
// mode 1: convT (Ci,Co,kh,kw), flipped:  B[k][co]=w[ci][co][kh-1-ky][kw-1-kx]
__global__ void k_packw(const float* __restrict__ w, _Float16* __restrict__ Bp,
                        int K, int Co, int Ci, int kh, int kw, int mode) {
  size_t t = (size_t)blockIdx.x * 256 + threadIdx.x;
  if (t >= (size_t)K * Co) return;
  int k  = (int)(t / Co);
  int co = (int)(t % Co);
  int ci = k % Ci;
  int tap = k / Ci;
  int ky = tap / kw, kx = tap % kw;
  float v;
  if (mode == 0)
    v = w[(((size_t)co * Ci + ci) * kh + ky) * kw + kx];
  else
    v = w[(((size_t)ci * Co + co) * kh + (kh - 1 - ky)) * kw + (kw - 1 - kx)];
  int kt = k >> 5, kin = k & 31, nt = co >> 4;
  int lane = (co & 15) + ((kin & 16) ? 16 : 0);
  int j = kin & 15;
  Bp[((size_t)(nt * (K >> 5) + kt) * 32 + lane) * 16 + j] = (_Float16)v;
}

// ---------------- elementwise: bias (+relu) ----------------
__global__ void k_bias_act(const float* __restrict__ xin, const float* __restrict__ bias,
                           float* __restrict__ out, size_t M, int C, int relu) {
  size_t i = (size_t)blockIdx.x * 256 + threadIdx.x;
  if (i >= M * (size_t)C) return;
  int c = (int)(i % C);
  float v = xin[i] + bias[c];
  if (relu) v = fmaxf(v, 0.0f);
  out[i] = v;
}

// ---------------- encoder conv1 direct (K=48, NCHW input) ----------------
__global__ void k_conv_e1(const float* __restrict__ x, const float* __restrict__ w,
                          const float* __restrict__ bias, float* __restrict__ y) {
  // x NCHW [64,3,128,128]; y NHWC [64,64,64,32]; k=4 s=2 p=1
  size_t t = (size_t)blockIdx.x * 256 + threadIdx.x;
  const size_t total = 64ull * 64 * 64 * 32;
  if (t >= total) return;
  int co = (int)(t % 32);
  int ox = (int)((t / 32) % 64);
  int oy = (int)((t / 2048) % 64);
  int b  = (int)(t / 131072);
  float acc = bias[co];
  for (int ky = 0; ky < 4; ++ky) {
    int iy = oy * 2 - 1 + ky;
    if (iy < 0 || iy >= 128) continue;
    for (int kx = 0; kx < 4; ++kx) {
      int ix = ox * 2 - 1 + kx;
      if (ix < 0 || ix >= 128) continue;
      for (int ci = 0; ci < 3; ++ci)
        acc += x[(((size_t)b * 3 + ci) * 128 + iy) * 128 + ix] *
               w[(((size_t)co * 3 + ci) * 4 + ky) * 4 + kx];
    }
  }
  y[t] = fmaxf(acc, 0.0f);
}

// ---------------- VQ helpers ----------------
__global__ void k_cnorm(const float* __restrict__ cb, float* __restrict__ cn) {
  int n = blockIdx.x * 256 + threadIdx.x;
  if (n >= 512) return;
  float s = 0.f;
  for (int c = 0; c < 64; ++c) { float v = cb[n * 64 + c]; s += v * v; }
  cn[n] = s;
}

__global__ void k_vq(const float* __restrict__ z, const float* __restrict__ dot,
                     const float* __restrict__ cb, const float* __restrict__ cn,
                     float* __restrict__ q, float* __restrict__ loss) {
  int m = blockIdx.x * 256 + threadIdx.x;
  if (m >= 65536) return;
  const float* zr = z + (size_t)m * 64;
  float zn = 0.f;
  for (int c = 0; c < 64; ++c) zn += zr[c] * zr[c];
  const float* dr = dot + (size_t)m * 512;
  float best = 3.4e38f; int bi = 0;
  for (int n = 0; n < 512; ++n) {
    float d = zn + cn[n] - 2.0f * dr[n];
    if (d < best) { best = d; bi = n; }
  }
  const float* cr = cb + (size_t)bi * 64;
  float* qr = q + (size_t)m * 64;
  float s = 0.f;
  for (int c = 0; c < 64; ++c) {
    float qc = cr[c];
    qr[c] = qc;
    float df = qc - zr[c];
    s += df * df;
  }
  atomicAdd(loss, s);
}

// ---------------- BatchNorm (training-mode batch stats) ----------------
__global__ void k_zero(float* __restrict__ p, int n) {
  int i = blockIdx.x * 256 + threadIdx.x;
  if (i < n) p[i] = 0.f;
}

__global__ void k_bn_stats(const float* __restrict__ x, float* __restrict__ sum,
                           float* __restrict__ sq, size_t M, int C) {
  __shared__ float ss[64], s2[64];
  if ((int)threadIdx.x < C) { ss[threadIdx.x] = 0.f; s2[threadIdx.x] = 0.f; }
  __syncthreads();
  const size_t total = M * (size_t)C;
  for (size_t i = (size_t)blockIdx.x * 256 + threadIdx.x; i < total;
       i += (size_t)gridDim.x * 256) {
    float v = x[i];
    int c = (int)(i % C);
    atomicAdd(&ss[c], v);
    atomicAdd(&s2[c], v * v);
  }
  __syncthreads();
  if ((int)threadIdx.x < C) {
    atomicAdd(&sum[threadIdx.x], ss[threadIdx.x]);
    atomicAdd(&sq[threadIdx.x],  s2[threadIdx.x]);
  }
}

__global__ void k_bn_apply(const float* __restrict__ xin, const float* __restrict__ res,
                           float* __restrict__ out, const float* __restrict__ g,
                           const float* __restrict__ be, const float* __restrict__ sum,
                           const float* __restrict__ sq, size_t M, int C,
                           float invCnt, int relu) {
  size_t i = (size_t)blockIdx.x * 256 + threadIdx.x;
  if (i >= M * (size_t)C) return;
  int c = (int)(i % C);
  float mean = sum[c] * invCnt;
  float var  = sq[c] * invCnt - mean * mean;
  float v = g[c] * (xin[i] - mean) * rsqrtf(var + 1e-5f) + be[c];
  if (res) v += res[i];
  if (relu) v = fmaxf(v, 0.0f);
  out[i] = v;
}

// ---------------- direct stride-2 transposed conv (NHWC) ----------------
__global__ void k_convt(const float* __restrict__ x, const float* __restrict__ w,
                        const float* __restrict__ bias, float* __restrict__ y,
                        int Bn, int IH, int IW, int Ci, int Co, int kk, int relu) {
  const int OH = IH * 2, OW = IW * 2;                   // stride 2, pad 1
  const size_t total = (size_t)Bn * OH * OW * Co;
  size_t t = (size_t)blockIdx.x * 256 + threadIdx.x;
  if (t >= total) return;
  int co = (int)(t % Co);
  int ox = (int)((t / Co) % OW);
  int oy = (int)((t / ((size_t)Co * OW)) % OH);
  int b  = (int)(t / ((size_t)Co * OW * OH));
  float acc = bias ? bias[co] : 0.0f;
  for (int ky = 0; ky < kk; ++ky) {
    int ty = oy + 1 - ky;
    if (ty & 1) continue;
    int iy = ty >> 1;
    if (iy < 0 || iy >= IH) continue;
    for (int kx = 0; kx < kk; ++kx) {
      int tx = ox + 1 - kx;
      if (tx & 1) continue;
      int ix = tx >> 1;
      if (ix < 0 || ix >= IW) continue;
      const float* xp = x + (((size_t)b * IH + iy) * IW + ix) * Ci;
      for (int ci = 0; ci < Ci; ++ci)
        acc += xp[ci] * w[(((size_t)ci * Co + co) * kk + ky) * kk + kx];
    }
  }
  if (relu) acc = fmaxf(acc, 0.0f);
  y[t] = acc;
}

// ---------------- final: BN(C=3) + sigmoid, NHWC -> NCHW, + loss ----------------
__global__ void k_final(const float* __restrict__ yraw, const float* __restrict__ g,
                        const float* __restrict__ be, const float* __restrict__ sum,
                        const float* __restrict__ sq, const float* __restrict__ loss,
                        float* __restrict__ out) {
  size_t t = (size_t)blockIdx.x * 256 + threadIdx.x;
  if (t == 0) out[3145728] = 1.25f * loss[0] / 4194304.0f;   // vq_loss
  const size_t total = 64ull * 3 * 128 * 128;
  if (t >= total) return;
  int hw = (int)(t % 16384);
  int c  = (int)((t / 16384) % 3);
  int b  = (int)(t / 49152);
  int yy = hw >> 7, xx = hw & 127;
  float v = yraw[(((size_t)b * 128 + yy) * 128 + xx) * 3 + c];
  const float invCnt = 1.0f / 1048576.0f;                    // N*H*W
  float mean = sum[c] * invCnt;
  float var  = sq[c] * invCnt - mean * mean;
  v = g[c] * (v - mean) * rsqrtf(var + 1e-5f) + be[c];
  out[t] = 1.0f / (1.0f + __expf(-v));
}

// =====================================================================
extern "C" void kernel_launch(void* const* d_in, const int* in_sizes, int n_in,
                              void* d_out, int out_size, void* d_ws, size_t ws_size,
                              hipStream_t stream) {
  (void)in_sizes; (void)n_in; (void)out_size; (void)ws_size;
  const float* x    = (const float*)d_in[0];
  const float* ew1  = (const float*)d_in[1];
  const float* eb1  = (const float*)d_in[2];
  const float* ew2  = (const float*)d_in[3];
  const float* eb2  = (const float*)d_in[4];
  const float* ew3  = (const float*)d_in[5];
  const float* eb3  = (const float*)d_in[6];
  const float* cb   = (const float*)d_in[7];
  const float* dw1  = (const float*)d_in[8];
  const float* db1  = (const float*)d_in[9];
  const float* r1w1 = (const float*)d_in[10];
  const float* r1g1 = (const float*)d_in[12];
  const float* r1be1= (const float*)d_in[13];
  const float* r1w2 = (const float*)d_in[14];
  const float* r1g2 = (const float*)d_in[16];
  const float* r1be2= (const float*)d_in[17];
  const float* r2w1 = (const float*)d_in[18];
  const float* r2g1 = (const float*)d_in[20];
  const float* r2be1= (const float*)d_in[21];
  const float* r2w2 = (const float*)d_in[22];
  const float* r2g2 = (const float*)d_in[24];
  const float* r2be2= (const float*)d_in[25];
  const float* dw2  = (const float*)d_in[26];
  const float* db2  = (const float*)d_in[27];
  const float* dw3  = (const float*)d_in[28];
  const float* bng  = (const float*)d_in[30];
  const float* bnb  = (const float*)d_in[31];
  float* out = (float*)d_out;

  char* ws = (char*)d_ws;
  const size_t OFF_A    = 0;                          // im2col A (<=151 MB)
  const size_t OFF_C    = 160ull << 20;               // GEMM out (<=134 MB)
  const size_t OFF_ACT1 = OFF_C    + (136ull << 20);
  const size_t OFF_ACT2 = OFF_ACT1 + (34ull  << 20);
  const size_t OFF_ACT3 = OFF_ACT2 + (34ull  << 20);
  const size_t OFF_ACT4 = OFF_ACT3 + (34ull  << 20);
  const size_t OFF_BP   = OFF_ACT4 + (34ull  << 20);  // packed weights (<=1 MB)
  const size_t OFF_CN   = OFF_BP   + (1ull   << 20);  // codebook norms
  const size_t OFF_BNS  = OFF_CN   + 4096;            // BN sum (64 f)
  const size_t OFF_BNQ  = OFF_BNS  + 256;             // BN sumsq (64 f)
  const size_t OFF_LOSS = OFF_BNQ  + 256;             // loss accumulator

  _Float16* A    = (_Float16*)(ws + OFF_A);
  float*    Cbuf = (float*)(ws + OFF_C);
  float*    act1 = (float*)(ws + OFF_ACT1);
  float*    act2 = (float*)(ws + OFF_ACT2);
  float*    act3 = (float*)(ws + OFF_ACT3);
  float*    act4 = (float*)(ws + OFF_ACT4);
  _Float16* Bp   = (_Float16*)(ws + OFF_BP);
  float*    cn   = (float*)(ws + OFF_CN);
  float*    bns  = (float*)(ws + OFF_BNS);
  float*    bnq  = (float*)(ws + OFF_BNQ);
  float*    loss = (float*)(ws + OFF_LOSS);

  auto blocks = [](size_t total) { return (unsigned)((total + 255) / 256); };

  // ---- encoder conv1 (direct, relu) -> act1 [64,64,64,32] ----
  k_conv_e1<<<blocks(8388608ull), 256, 0, stream>>>(x, ew1, eb1, act1);

  // ---- e2: 4x4 s2 p1, K=512, N=64 -> act2 [64,32,32,64] ----
  k_packw<<<blocks(512ull * 64), 256, 0, stream>>>(ew2, Bp, 512, 64, 32, 4, 4, 0);
  k_im2col<<<blocks(65536ull * 16), 256, 0, stream>>>(act1, A, 64, 64, 64, 32, 4, 4, 2, 1, 32, 32);
  k_gemm_wmma_n64<<<dim3(512, 1), 256, 0, stream>>>(A, Bp, Cbuf, 65536, 64, 512);
  k_bias_act<<<blocks(65536ull * 64), 256, 0, stream>>>(Cbuf, eb2, act2, 65536, 64, 1);

  // ---- e3: 3x3 s1 p1, K=576, N=64 -> z = act1 [64,32,32,64] ----
  k_packw<<<blocks(576ull * 64), 256, 0, stream>>>(ew3, Bp, 576, 64, 64, 3, 3, 0);
  k_im2col<<<blocks(65536ull * 9), 256, 0, stream>>>(act2, A, 64, 32, 32, 64, 3, 3, 1, 1, 32, 32);
  k_gemm_wmma_n64<<<dim3(512, 1), 256, 0, stream>>>(A, Bp, Cbuf, 65536, 64, 576);
  k_bias_act<<<blocks(65536ull * 64), 256, 0, stream>>>(Cbuf, eb3, act1, 65536, 64, 0);

  // ---- VQ: dot = z @ cb^T via WMMA (M=65536,N=512,K=64), then argmin -> q=act2 ----
  k_packw<<<blocks(64ull * 512), 256, 0, stream>>>(cb, Bp, 64, 512, 64, 1, 1, 0);
  k_im2col<<<blocks(65536ull), 256, 0, stream>>>(act1, A, 64, 32, 32, 64, 1, 1, 1, 0, 32, 32);
  k_gemm_wmma_n64<<<dim3(512, 8), 256, 0, stream>>>(A, Bp, Cbuf, 65536, 512, 64);
  k_cnorm<<<2, 256, 0, stream>>>(cb, cn);
  k_zero<<<1, 256, 0, stream>>>(loss, 1);
  k_vq<<<256, 256, 0, stream>>>(act1, Cbuf, cb, cn, act2, loss);

  // ---- dconv1: convT k3 s1 p1 == conv(flipped), relu -> y1 = act3 ----
  k_packw<<<blocks(576ull * 64), 256, 0, stream>>>(dw1, Bp, 576, 64, 64, 3, 3, 1);
  k_im2col<<<blocks(65536ull * 9), 256, 0, stream>>>(act2, A, 64, 32, 32, 64, 3, 3, 1, 1, 32, 32);
  k_gemm_wmma_n64<<<dim3(512, 1), 256, 0, stream>>>(A, Bp, Cbuf, 65536, 64, 576);
  k_bias_act<<<blocks(65536ull * 64), 256, 0, stream>>>(Cbuf, db1, act3, 65536, 64, 1);

  // ---- resblock 1, conv1 + BN + relu -> act2 ----
  k_packw<<<blocks(576ull * 64), 256, 0, stream>>>(r1w1, Bp, 576, 64, 64, 3, 3, 0);
  k_im2col<<<blocks(65536ull * 9), 256, 0, stream>>>(act3, A, 64, 32, 32, 64, 3, 3, 1, 1, 32, 32);
  k_gemm_wmma_n64<<<dim3(512, 1), 256, 0, stream>>>(A, Bp, Cbuf, 65536, 64, 576);
  k_zero<<<1, 256, 0, stream>>>(bns, 128);
  k_bn_stats<<<1024, 256, 0, stream>>>(Cbuf, bns, bnq, 65536, 64);
  k_bn_apply<<<blocks(65536ull * 64), 256, 0, stream>>>(Cbuf, nullptr, act2, r1g1, r1be1,
                                                        bns, bnq, 65536, 64, 1.0f / 65536.0f, 1);
  // ---- resblock 1, conv2 + BN + skip(y1) + relu -> act1 ----
  k_packw<<<blocks(576ull * 64), 256, 0, stream>>>(r1w2, Bp, 576, 64, 64, 3, 3, 0);
  k_im2col<<<blocks(65536ull * 9), 256, 0, stream>>>(act2, A, 64, 32, 32, 64, 3, 3, 1, 1, 32, 32);
  k_gemm_wmma_n64<<<dim3(512, 1), 256, 0, stream>>>(A, Bp, Cbuf, 65536, 64, 576);
  k_zero<<<1, 256, 0, stream>>>(bns, 128);
  k_bn_stats<<<1024, 256, 0, stream>>>(Cbuf, bns, bnq, 65536, 64);
  k_bn_apply<<<blocks(65536ull * 64), 256, 0, stream>>>(Cbuf, act3, act1, r1g2, r1be2,
                                                        bns, bnq, 65536, 64, 1.0f / 65536.0f, 1);

  // ---- dconv2: convT k4 s2 p1 (direct) + relu -> y2 = act4 [64,64,64,32] ----
  k_convt<<<blocks(8388608ull), 256, 0, stream>>>(act1, dw2, db2, act4, 64, 32, 32, 64, 32, 4, 1);

  // ---- resblock 2, conv1 + BN + relu -> act1 (M=262144,K=288,N=32) ----
  k_packw<<<blocks(288ull * 32), 256, 0, stream>>>(r2w1, Bp, 288, 32, 32, 3, 3, 0);
  k_im2col<<<blocks(262144ull * 9), 256, 0, stream>>>(act4, A, 64, 64, 64, 32, 3, 3, 1, 1, 64, 64);
  k_gemm_wmma<<<dim3(2048, 1), 256, 0, stream>>>(A, Bp, Cbuf, 262144, 32, 288);
  k_zero<<<1, 256, 0, stream>>>(bns, 128);
  k_bn_stats<<<1024, 256, 0, stream>>>(Cbuf, bns, bnq, 262144, 32);
  k_bn_apply<<<blocks(262144ull * 32), 256, 0, stream>>>(Cbuf, nullptr, act1, r2g1, r2be1,
                                                         bns, bnq, 262144, 32, 1.0f / 262144.0f, 1);
  // ---- resblock 2, conv2 + BN + skip(y2) + relu -> act3 ----
  k_packw<<<blocks(288ull * 32), 256, 0, stream>>>(r2w2, Bp, 288, 32, 32, 3, 3, 0);
  k_im2col<<<blocks(262144ull * 9), 256, 0, stream>>>(act1, A, 64, 64, 64, 32, 3, 3, 1, 1, 64, 64);
  k_gemm_wmma<<<dim3(2048, 1), 256, 0, stream>>>(A, Bp, Cbuf, 262144, 32, 288);
  k_zero<<<1, 256, 0, stream>>>(bns, 128);
  k_bn_stats<<<1024, 256, 0, stream>>>(Cbuf, bns, bnq, 262144, 32);
  k_bn_apply<<<blocks(262144ull * 32), 256, 0, stream>>>(Cbuf, act4, act3, r2g2, r2be2,
                                                         bns, bnq, 262144, 32, 1.0f / 262144.0f, 1);

  // ---- dconv3: convT k4 s2 p1 (direct, no bias: cancels in BN) -> act2 [64,128,128,3] ----
  k_convt<<<blocks(3145728ull), 256, 0, stream>>>(act3, dw3, nullptr, act2, 64, 64, 64, 32, 3, 4, 0);

  // ---- final BN (C=3) + sigmoid, NHWC->NCHW, append vq_loss ----
  k_zero<<<1, 256, 0, stream>>>(bns, 128);
  k_bn_stats<<<1024, 256, 0, stream>>>(act2, bns, bnq, 1048576, 3);
  k_final<<<blocks(3145728ull), 256, 0, stream>>>(act2, bng, bnb, bns, bnq, loss, out);
}
